// T5DecoderHead_68298569941243
// MI455X (gfx1250) — matrix operand order repaired and checked by reading, not compiled, not run on hardware
//
#include <hip/hip_runtime.h>
#include <math.h>

#define D_MODEL 768
#define VOCAB   32128
#define NROWS   2048      // B*S = 4*512
#define IGNORE_INDEX (-100)
#define LN_EPS  1e-6f

// ---- GEMM tiling: block 256 threads = 8 waves (4 M x 2 N), wave tile 64x64 ----
#define GBM 256           // block M tile  (grid.x = 2048/256 = 8)
#define GBN 128           // block N tile  (grid.y = 32128/128 = 251)
#define GKC 64            // K chunk staged in LDS
#define LDH 72            // padded LDS row stride in halves: 144B -> bank 36*r%64
                          // distinct for 16 consecutive rows, and 16B aligned

typedef _Float16 half_t;
typedef __attribute__((ext_vector_type(16))) _Float16     v16h;
typedef __attribute__((ext_vector_type(8)))  float        v8f;
typedef __attribute__((ext_vector_type(4)))  unsigned int v4u;

union Frag16 { v16h v; v4u q[2]; };

// ---------------------------------------------------------------------------
// 1) T5 RMSNorm -> f16 activations:  h16 = (f16)(ln_w * x * rsqrt(mean(x^2)+eps))
// ---------------------------------------------------------------------------
__global__ __launch_bounds__(256)
void rmsnorm_kernel(const float* __restrict__ x,
                    const float* __restrict__ w,
                    half_t* __restrict__ h16) {
    const int row = blockIdx.x;
    const int t   = threadIdx.x;
    const float* xr = x + (size_t)row * D_MODEL;

    float v0 = xr[t];
    float v1 = xr[t + 256];
    float v2 = xr[t + 512];

    __shared__ float red[256];
    red[t] = v0 * v0 + v1 * v1 + v2 * v2;
    __syncthreads();
    #pragma unroll
    for (int off = 128; off > 0; off >>= 1) {
        if (t < off) red[t] += red[t + off];
        __syncthreads();
    }
    const float inv = rsqrtf(red[0] * (1.0f / (float)D_MODEL) + LN_EPS);

    half_t* hr = h16 + (size_t)row * D_MODEL;
    hr[t]       = (half_t)(w[t]       * v0 * inv);
    hr[t + 256] = (half_t)(w[t + 256] * v1 * inv);
    hr[t + 512] = (half_t)(w[t + 512] * v2 * inv);
}

// ---------------------------------------------------------------------------
// 2a) Optional one-shot weight conversion fp32 -> fp16 (L2/HBM streaming)
// ---------------------------------------------------------------------------
__global__ __launch_bounds__(256)
void w_to_f16_kernel(const float* __restrict__ src, half_t* __restrict__ dst) {
    const size_t total16 = (size_t)VOCAB * D_MODEL / 16;
    for (size_t i = (size_t)blockIdx.x * 256 + threadIdx.x; i < total16;
         i += (size_t)gridDim.x * 256) {
        const float* s = src + i * 16;
        Frag16 f;
        #pragma unroll
        for (int j = 0; j < 16; ++j) f.v[j] = (half_t)s[j];
        v4u* d = (v4u*)(dst + i * 16);
        d[0] = f.q[0];
        d[1] = f.q[1];
    }
}

// ---------------------------------------------------------------------------
// 2b) Logits GEMM via V_WMMA_F32_16X16X32_F16, fp32 accumulate.
//     C[m, v] = sum_k h[m,k] * W[v,k]
//     Block tile 256x128, both operands staged in LDS (54KB), K chunks of 64.
//     WF16: weights already f16 in workspace; else convert f32->f16 inline.
// ---------------------------------------------------------------------------
template <bool WF16>
__global__ __launch_bounds__(256)
void gemm_wmma_f16_kernel(const half_t* __restrict__ A16,  // [NROWS][D_MODEL]
                          const half_t* __restrict__ W16,  // [VOCAB][D_MODEL] if WF16
                          const float*  __restrict__ W32,  // [VOCAB][D_MODEL] if !WF16
                          float* __restrict__ C) {         // [NROWS][VOCAB]
    __shared__ half_t As[GBM * LDH];   // 36864 B
    __shared__ half_t Bs[GBN * LDH];   // 18432 B

    const int m0   = blockIdx.x * GBM;
    const int n0   = blockIdx.y * GBN;
    const int t    = threadIdx.x;     // 0..255
    const int wave = t >> 5;          // 0..7
    const int lane = t & 31;
    const int hi   = lane >> 4;       // half-wave select
    const int lidx = lane & 15;
    const int wm   = wave >> 1;       // 0..3 : wave M tile (64 rows each)
    const int wn   = wave & 1;        // 0..1 : wave N tile (64 cols each)

    v8f acc[4][4];
    #pragma unroll
    for (int mi = 0; mi < 4; ++mi)
        #pragma unroll
        for (int ni = 0; ni < 4; ++ni)
            #pragma unroll
            for (int j = 0; j < 8; ++j) acc[mi][ni][j] = 0.0f;

    for (int kc = 0; kc < D_MODEL; kc += GKC) {
        __syncthreads();
        // A chunk: 256 rows x 64 halves = 2048 vec8 -> 8 per thread
        #pragma unroll
        for (int f = t; f < GBM * (GKC / 8); f += 256) {
            const int r = f >> 3, c8 = f & 7;
            v4u q = *(const v4u*)(A16 + (size_t)(m0 + r) * D_MODEL + kc + c8 * 8);
            *(v4u*)(&As[r * LDH + c8 * 8]) = q;
        }
        // B chunk: 128 rows x 64 halves = 1024 vec8 -> 4 per thread
        #pragma unroll
        for (int f = t; f < GBN * (GKC / 8); f += 256) {
            const int r = f >> 3, c8 = f & 7;
            if (WF16) {
                v4u q = *(const v4u*)(W16 + (size_t)(n0 + r) * D_MODEL + kc + c8 * 8);
                *(v4u*)(&Bs[r * LDH + c8 * 8]) = q;
            } else {
                const float* s = W32 + (size_t)(n0 + r) * D_MODEL + kc + c8 * 8;
                Frag16 fq;
                #pragma unroll
                for (int j = 0; j < 8; ++j) fq.v[j] = (half_t)s[j];
                *(v4u*)(&Bs[r * LDH + c8 * 8]) = fq.q[0];
            }
        }
        __syncthreads();

        #pragma unroll
        for (int kk = 0; kk < GKC; kk += 32) {
            Frag16 a[4], b[4];
            // A 16x32 f16 layout: halves 0-7 = K(hi*8..), halves 8-15 = K(16+hi*8..)
            #pragma unroll
            for (int mi = 0; mi < 4; ++mi) {
                const half_t* p = &As[(wm * 64 + mi * 16 + lidx) * LDH + kk];
                a[mi].q[0] = *(const v4u*)(p + hi * 8);
                a[mi].q[1] = *(const v4u*)(p + 16 + hi * 8);
            }
            // B 32x16 f16 layout: halves 0-7 = K(hi*8..), halves 8-15 = K(16+hi*8..)
            #pragma unroll
            for (int ni = 0; ni < 4; ++ni) {
                const half_t* p = &Bs[(wn * 64 + ni * 16 + lidx) * LDH + kk];
                b[ni].q[0] = *(const v4u*)(p + hi * 8);
                b[ni].q[1] = *(const v4u*)(p + 16 + hi * 8);
            }
            #pragma unroll
            for (int mi = 0; mi < 4; ++mi)
                #pragma unroll
                for (int ni = 0; ni < 4; ++ni)
                    acc[mi][ni] = __builtin_amdgcn_wmma_f32_16x16x32_f16(
                        false, a[mi].v, false, b[ni].v,
                        (short)0, acc[mi][ni], false, false);
        }
    }

    // C/D layout: VGPR r -> M = r (lanes 0-15) / r+8 (lanes 16-31), N = lane&15
    #pragma unroll
    for (int mi = 0; mi < 4; ++mi) {
        #pragma unroll
        for (int r = 0; r < 8; ++r) {
            const int m = m0 + wm * 64 + mi * 16 + r + hi * 8;
            float* crow = C + (size_t)m * VOCAB + n0 + wn * 64 + lidx;
            #pragma unroll
            for (int ni = 0; ni < 4; ++ni) crow[ni * 16] = acc[mi][ni][r];
        }
    }
}

// ---------------------------------------------------------------------------
// 3) Per-row online log-softmax + NLL (single pass over 32128 logits)
// ---------------------------------------------------------------------------
__global__ __launch_bounds__(256)
void ce_rowstats_kernel(const float* __restrict__ logits,
                        const int* __restrict__ labels, int lab_stride,
                        float* __restrict__ nll,
                        float* __restrict__ validf) {
    const int row = blockIdx.x;
    const float* lr = logits + (size_t)row * VOCAB;

    float m = -INFINITY, s = 0.0f;
    for (int i = threadIdx.x; i < VOCAB; i += 256) {
        const float xv = lr[i];
        const float mn = fmaxf(m, xv);
        s = s * __expf(m - mn) + __expf(xv - mn);
        m = mn;
    }

    __shared__ float sm[256], ss[256];
    sm[threadIdx.x] = m;
    ss[threadIdx.x] = s;
    __syncthreads();
    #pragma unroll
    for (int off = 128; off > 0; off >>= 1) {
        if (threadIdx.x < off) {
            const float m1 = sm[threadIdx.x], s1 = ss[threadIdx.x];
            const float m2 = sm[threadIdx.x + off], s2 = ss[threadIdx.x + off];
            const float mn = fmaxf(m1, m2);
            ss[threadIdx.x] = s1 * __expf(m1 - mn) + s2 * __expf(m2 - mn);
            sm[threadIdx.x] = mn;
        }
        __syncthreads();
    }

    if (threadIdx.x == 0) {
        // low 32 bits correct for both int32 and little-endian int64 labels
        const int lab = labels[(size_t)row * lab_stride];
        const bool valid = (lab != IGNORE_INDEX);
        const float lse = sm[0] + __logf(ss[0]);
        nll[row]    = valid ? (lse - lr[lab]) : 0.0f;
        validf[row] = valid ? 1.0f : 0.0f;
    }
}

// ---------------------------------------------------------------------------
// 4) Deterministic final reduction: loss = sum(nll) / max(sum(valid), 1)
// ---------------------------------------------------------------------------
__global__ __launch_bounds__(256)
void ce_finalize_kernel(const float* __restrict__ nll,
                        const float* __restrict__ validf,
                        float* __restrict__ out_loss) {
    __shared__ float sn[256], sv[256];
    float a = 0.0f, b = 0.0f;
    for (int i = threadIdx.x; i < NROWS; i += 256) {
        a += nll[i];
        b += validf[i];
    }
    sn[threadIdx.x] = a;
    sv[threadIdx.x] = b;
    __syncthreads();
    #pragma unroll
    for (int off = 128; off > 0; off >>= 1) {
        if (threadIdx.x < off) {
            sn[threadIdx.x] += sn[threadIdx.x + off];
            sv[threadIdx.x] += sv[threadIdx.x + off];
        }
        __syncthreads();
    }
    if (threadIdx.x == 0) out_loss[0] = sn[0] / fmaxf(sv[0], 1.0f);
}

// ---------------------------------------------------------------------------
extern "C" void kernel_launch(void* const* d_in, const int* in_sizes, int n_in,
                              void* d_out, int out_size, void* d_ws, size_t ws_size,
                              hipStream_t stream) {
    const float* hidden = (const float*)d_in[0];   // [4,512,768] fp32
    const int*   labels = (const int*)d_in[1];     // [4,512] (stride hedge below)
    const float* ln_w   = (const float*)d_in[2];   // [768]
    const float* lm_w   = (const float*)d_in[3];   // [32128,768] fp32

    float* out    = (float*)d_out;
    float* loss   = out;          // output[0]  : scalar loss
    float* logits = out + 1;      // output[1..]: [2048][32128] prediction_scores

    // Workspace layout
    const size_t h16_bytes   = (size_t)NROWS * D_MODEL * sizeof(half_t);  // 3.1 MB
    const size_t stats_bytes = (size_t)2 * NROWS * sizeof(float);         // 16 KB
    const size_t w16_bytes   = (size_t)VOCAB * D_MODEL * sizeof(half_t);  // 49.3 MB

    char*   ws     = (char*)d_ws;
    half_t* h16    = (half_t*)ws;
    float*  nll    = (float*)(ws + h16_bytes);
    float*  validf = nll + NROWS;
    half_t* w16    = (half_t*)(ws + h16_bytes + stats_bytes);

    const bool use_w16 = (ws_size >= h16_bytes + stats_bytes + w16_bytes);
    const int  lab_stride = (in_sizes[1] == 2 * NROWS) ? 2 : 1;  // int64 vs int32

    rmsnorm_kernel<<<NROWS, 256, 0, stream>>>(hidden, ln_w, h16);

    dim3 grid(NROWS / GBM, VOCAB / GBN);   // 8 x 251
    if (use_w16) {
        w_to_f16_kernel<<<4096, 256, 0, stream>>>(lm_w, w16);
        gemm_wmma_f16_kernel<true><<<grid, 256, 0, stream>>>(h16, w16, nullptr, logits);
    } else {
        gemm_wmma_f16_kernel<false><<<grid, 256, 0, stream>>>(h16, nullptr, lm_w, logits);
    }

    ce_rowstats_kernel<<<NROWS, 256, 0, stream>>>(logits, labels, lab_stride, nll, validf);
    ce_finalize_kernel<<<1, 256, 0, stream>>>(nll, validf, loss);
}
